// SpectralGatingNetwork_27986006900974
// MI455X (gfx1250) — compile-verified, matrix operand order checked by reading, not run on hardware
//
#include <hip/hip_runtime.h>
#include <hip/hip_bf16.h>
#include <math.h>

// ---------------------------------------------------------------- constants
#define B_    2
#define T_    2048
#define V_    32000
#define D_    1024
#define F_    256
#define NL_   6
#define LAT_  64
#define M_    (B_ * T_)          // 4096 rows
#define LDSTR 48                 // padded LDS row stride (bf16), 96B = 16B-aligned

#ifndef USE_ASYNC_LDS
#define USE_ASYNC_LDS 1          // CDNA5 global_load_async_to_lds_b128 path
#endif

typedef unsigned short u16;
typedef __attribute__((ext_vector_type(16))) __bf16 v16bf;
typedef __attribute__((ext_vector_type(8)))  float  v8f;

// fp32 -> bf16, round-to-nearest-even
__device__ __forceinline__ u16 f2bf(float f) {
    unsigned int u = __float_as_uint(f);
    u += 0x7FFFu + ((u >> 16) & 1u);
    return (u16)(u >> 16);
}

// ---------------------------------------------------------------- staging
// Copy 16B global -> LDS. Async path uses the CDNA5 async-DMA (ASYNCcnt).
__device__ __forceinline__ void stage16(u16* ldsDst, const u16* gSrc) {
#if USE_ASYNC_LDS
    // LDS byte address = low 32 bits of the generic pointer (ISA aperture rule)
    unsigned lds = (unsigned)(uintptr_t)ldsDst;
    asm volatile("global_load_async_to_lds_b128 %0, %1, off"
                 :: "v"(lds), "v"(gSrc) : "memory");
#else
    *(uint4*)ldsDst = *(const uint4*)gSrc;
#endif
}

#if USE_ASYNC_LDS
#define WAIT_ASYNC(n) asm volatile("s_wait_asynccnt %0" :: "n"(n) : "memory")
#else
#define WAIT_ASYNC(n)
#endif

// Stage one 128x32 bf16 tile (512 16B chunks, 2 per thread)
__device__ __forceinline__ void stage_tile(const u16* __restrict__ g, int ldg,
                                           u16* ldsBuf, int tid) {
    #pragma unroll
    for (int j = 0; j < 2; ++j) {
        int c  = tid + j * 256;          // 0..511
        int r  = c >> 2;                 // row 0..127
        int c8 = (c & 3) * 8;            // bf16 col 0/8/16/24
        stage16(ldsBuf + r * LDSTR + c8, g + (size_t)r * ldg + c8);
    }
}

// Load one 16x32 bf16 fragment from LDS per the CDNA5 WMMA layout:
// lanes 0-15 : row = rowBase+lane,    K = {0..7, 16..23}
// lanes 16-31: row = rowBase+lane-16, K = {8..15, 24..31}
__device__ __forceinline__ v16bf load_frag(const u16* lds, int rowBase, int lane) {
    const int r  = rowBase + (lane & 15);
    const int kh = (lane >> 4) << 3;     // 0 or 8
    union { uint4 q[2]; v16bf v; } u;
    u.q[0] = *reinterpret_cast<const uint4*>(lds + r * LDSTR + kh);
    u.q[1] = *reinterpret_cast<const uint4*>(lds + r * LDSTR + kh + 16);
    return u.v;
}

// ---------------------------------------------------------------- WMMA GEMM
// out[M,N] = epi( A[M,K](bf16, lda) @ W[N,K](bf16)^T + bias ) * mul + add
// act: 0=none, 1=sigmoid, 2=exact gelu. bias/mul/add/out/outbf nullable.
// ntStore: non-temporal hint on out stores (streaming outputs, keep L2 for weights).
// Requires M%128==0, N%128==0, K%32==0 (true for all calls here).
__global__ __launch_bounds__(256) void gemm_bf16_wmma(
    const u16* __restrict__ A, int lda,
    const u16* __restrict__ W,
    const float* __restrict__ bias,
    const float* __restrict__ mul,
    const float* __restrict__ add,
    float* __restrict__ out,
    u16*   __restrict__ outbf,
    int N, int K, int act, int ntStore)
{
    __shared__ u16 ldsA[2][128 * LDSTR];   // 2 x 12KB
    __shared__ u16 ldsB[2][128 * LDSTR];   // 2 x 12KB

    const int tid   = threadIdx.x;
    const int lane  = tid & 31;
    const int wave  = tid >> 5;
    const int waveM = wave >> 2;          // 0..1 -> 64-row strip
    const int waveN = wave & 3;           // 0..3 -> 32-col strip
    const int blockM = blockIdx.y * 128;
    const int blockN = blockIdx.x * 128;

    const u16* Ab = A + (size_t)blockM * lda;
    const u16* Wb = W + (size_t)blockN * K;

    v8f acc[4][2] = {};
    const int numK = K >> 5;

    // prologue: stage tile 0 into buffer 0
    stage_tile(Ab, lda, ldsA[0], tid);
    stage_tile(Wb, K,   ldsB[0], tid);

    for (int kt = 0; kt < numK; ++kt) {
        const int cur = kt & 1;
        if (kt + 1 < numK) {
            // overlap: async engine fills next buffer during this tile's math
            stage_tile(Ab + ((kt + 1) << 5), lda, ldsA[cur ^ 1], tid);
            stage_tile(Wb + ((kt + 1) << 5), K,   ldsB[cur ^ 1], tid);
            WAIT_ASYNC(4);     // own 4 ops for tile kt retired (in-order)
        } else {
            WAIT_ASYNC(0);
        }
        __syncthreads();       // tile kt visible to all waves

        v16bf afrag[4], bfrag[2];
        #pragma unroll
        for (int t = 0; t < 4; ++t)
            afrag[t] = load_frag(ldsA[cur], waveM * 64 + t * 16, lane);
        #pragma unroll
        for (int t = 0; t < 2; ++t)
            bfrag[t] = load_frag(ldsB[cur], waveN * 32 + t * 16, lane);

        #pragma unroll
        for (int tm = 0; tm < 4; ++tm)
            #pragma unroll
            for (int tn = 0; tn < 2; ++tn)
                acc[tm][tn] = __builtin_amdgcn_wmma_f32_16x16x32_bf16(
                    false, afrag[tm], false, bfrag[tn],
                    (short)0, acc[tm][tn], false, false);

        __syncthreads();       // done reading buf[cur] before it is refilled
    }

    // epilogue — C/D layout: lane<16 -> N=lane, M=r; lane>=16 -> N=lane-16, M=r+8
    const int colL = lane & 15;
    const int rowL = (lane >> 4) * 8;
    #pragma unroll
    for (int tm = 0; tm < 4; ++tm) {
        #pragma unroll
        for (int tn = 0; tn < 2; ++tn) {
            const int col = blockN + waveN * 32 + tn * 16 + colL;
            const float bv = bias ? bias[col] : 0.0f;
            #pragma unroll
            for (int r = 0; r < 8; ++r) {
                const int row = blockM + waveM * 64 + tm * 16 + rowL + r;
                const size_t idx = (size_t)row * N + col;
                float z = acc[tm][tn][r] + bv;
                if (act == 1)      z = 1.0f / (1.0f + expf(-z));
                else if (act == 2) z = 0.5f * z * (1.0f + erff(z * 0.70710678118654752f));
                if (mul) z *= mul[idx];
                if (add) z += add[idx];
                if (out) {
                    if (ntStore) __builtin_nontemporal_store(z, &out[idx]);
                    else         out[idx] = z;
                }
                if (outbf) outbf[idx] = f2bf(z);
            }
        }
    }
}

// ------------------------------------------------------- fp32 -> bf16 convert
__global__ __launch_bounds__(256) void cvt_bf16_k(
    const float* __restrict__ src, u16* __restrict__ dst, int n4)
{
    const int i = blockIdx.x * blockDim.x + threadIdx.x;
    if (i >= n4) return;
    const float4 v = *(const float4*)(src + (size_t)i * 4);
    const unsigned p0 = (unsigned)f2bf(v.x) | ((unsigned)f2bf(v.y) << 16);
    const unsigned p1 = (unsigned)f2bf(v.z) | ((unsigned)f2bf(v.w) << 16);
    *(uint2*)(dst + (size_t)i * 4) = make_uint2(p0, p1);
}

// ---------------------------------------------------------------- LayerNorm
__global__ __launch_bounds__(256) void layernorm_k(
    const float* __restrict__ x, const float* __restrict__ g,
    const float* __restrict__ b, float* __restrict__ out,
    u16* __restrict__ outbf)
{
    __shared__ float s1[256], s2[256];
    const int row = blockIdx.x;
    const int t   = threadIdx.x;
    const float* xr = x + (size_t)row * D_;
    float v[4]; float a = 0.f, q = 0.f;
    #pragma unroll
    for (int j = 0; j < 4; ++j) { v[j] = xr[t + j * 256]; a += v[j]; q += v[j] * v[j]; }
    s1[t] = a; s2[t] = q; __syncthreads();
    for (int off = 128; off > 0; off >>= 1) {
        if (t < off) { s1[t] += s1[t + off]; s2[t] += s2[t + off]; }
        __syncthreads();
    }
    const float mean = s1[0] * (1.0f / D_);
    const float var  = s2[0] * (1.0f / D_) - mean * mean;
    const float inv  = rsqrtf(var + 1e-5f);
    #pragma unroll
    for (int j = 0; j < 4; ++j) {
        const int c = t + j * 256;
        const float o = (v[j] - mean) * inv * g[c] + b[c];
        if (out)   out[(size_t)row * D_ + c]   = o;
        outbf[(size_t)row * D_ + c] = f2bf(o);
    }
}

// ---------------------------------------------------------------- embedding
__global__ __launch_bounds__(256) void embed_k(
    const int* __restrict__ x, const float* __restrict__ emb, float* __restrict__ h)
{
    const int row = blockIdx.x;
    const int tok = x[row];
    const float4* src = (const float4*)(emb + (size_t)tok * D_);
    float4* dst = (float4*)(h + (size_t)row * D_);
    dst[threadIdx.x] = src[threadIdx.x];   // 256 * 4 = 1024 = D_
}

// -------------------------------------------------- complex linear scan (T)
// y_t = a_t * y_{t-1} + u_t  (complex), inclusive, per (b,f) channel
__global__ void scan_k(const float* __restrict__ decay,
                       const float* __restrict__ sp,
                       const float* __restrict__ freq,
                       u16* __restrict__ yc)
{
    const int c = blockIdx.x * blockDim.x + threadIdx.x;
    if (c >= B_ * F_) return;
    const int b = c >> 8;            // / F_
    const int f = c & (F_ - 1);
    const float w  = freq[f] * 0.1f;
    const float cw = cosf(w), sw = sinf(w);
    float yr = 0.f, yi = 0.f;
    const size_t base = (size_t)b * T_;
    for (int t = 0; t < T_; ++t) {
        const size_t m = base + t;
        const float d  = decay[m * F_ + f];
        const float ar = d * cw, ai = d * sw;
        const float ur = sp[m * (2 * F_) + f];
        const float ui = sp[m * (2 * F_) + F_ + f];
        const float nyr = ar * yr - ai * yi + ur;
        const float nyi = ar * yi + ai * yr + ui;
        yr = nyr; yi = nyi;
        yc[m * (2 * F_) + f]      = f2bf(yr);
        yc[m * (2 * F_) + F_ + f] = f2bf(yi);
    }
}

// ---------------------------------------------------------------- KL loss
__global__ void init_acc_k(float* acc) { acc[0] = 0.0f; }

__global__ __launch_bounds__(256) void kl_reduce_k(
    const float* __restrict__ params, float* __restrict__ acc)
{
    __shared__ float s[256];
    float partial = 0.f;
    const int total = M_ * LAT_;
    for (int i = blockIdx.x * blockDim.x + threadIdx.x; i < total;
         i += gridDim.x * blockDim.x) {
        const int row = i >> 6;
        const int j   = i & 63;
        const float mu = params[row * (2 * LAT_) + j];
        const float lv = params[row * (2 * LAT_) + LAT_ + j];
        partial += -0.5f * (1.0f + lv - mu * mu - expf(lv));
    }
    s[threadIdx.x] = partial; __syncthreads();
    for (int off = 128; off > 0; off >>= 1) {
        if (threadIdx.x < off) s[threadIdx.x] += s[threadIdx.x + off];
        __syncthreads();
    }
    if (threadIdx.x == 0) atomicAdd(acc, s[0]);
}

__global__ void kl_final_k(const float* __restrict__ acc, float* __restrict__ out) {
    out[0] = 0.01f * fmaxf(acc[0] * (1.0f / (float)(M_ * LAT_)), 0.05f);
}

// ---------------------------------------------------------------- host side
static inline void launch_gemm(const u16* A, int lda, const u16* W,
                               const float* bias, const float* mul,
                               const float* add, float* out, u16* outbf,
                               int N, int K, int act, int nt, hipStream_t s)
{
    dim3 grid(N / 128, M_ / 128);
    gemm_bf16_wmma<<<grid, 256, 0, s>>>(A, lda, W, bias, mul, add, out, outbf,
                                        N, K, act, nt);
}

static inline void launch_cvt(const float* src, u16* dst, size_t n, hipStream_t s)
{
    const int n4 = (int)(n / 4);
    cvt_bf16_k<<<(n4 + 255) / 256, 256, 0, s>>>(src, dst, n4);
}

extern "C" void kernel_launch(void* const* d_in, const int* in_sizes, int n_in,
                              void* d_out, int out_size, void* d_ws, size_t ws_size,
                              hipStream_t stream)
{
    (void)in_sizes; (void)n_in; (void)out_size; (void)ws_size;

    const int*   x          = (const int*)  d_in[0];
    const float* emb        = (const float*)d_in[1];
    const float* freq       = (const float*)d_in[2];
    const float* decay_W    = (const float*)d_in[3];
    const float* decay_b    = (const float*)d_in[4];
    const float* spec_in_W  = (const float*)d_in[5];
    const float* spec_out_W = (const float*)d_in[6];
    const float* norm1_g    = (const float*)d_in[7];
    const float* norm1_b    = (const float*)d_in[8];
    const float* gate_W     = (const float*)d_in[9];
    const float* gate_b     = (const float*)d_in[10];
    const float* ogate_W    = (const float*)d_in[11];
    const float* ogate_b    = (const float*)d_in[12];
    const float* ffn_g      = (const float*)d_in[13];
    const float* ffn_b      = (const float*)d_in[14];
    const float* ffn_W1     = (const float*)d_in[15];
    const float* ffn_b1     = (const float*)d_in[16];
    const float* ffn_W2     = (const float*)d_in[17];
    const float* ffn_b2     = (const float*)d_in[18];
    const float* vib_g      = (const float*)d_in[19];
    const float* vib_b      = (const float*)d_in[20];
    const float* enc_W      = (const float*)d_in[21];
    const float* enc_b      = (const float*)d_in[22];
    const float* dec_W      = (const float*)d_in[23];
    const float* dec_b      = (const float*)d_in[24];
    const float* normf_g    = (const float*)d_in[25];
    const float* normf_b    = (const float*)d_in[26];

    float* logits = (float*)d_out;                  // [M_, V_]
    float* reg    = logits + (size_t)M_ * V_;       // scalar

    // -------- scratch layout
    float* ws  = (float*)d_ws;
    float* h   = ws;                                // M*D
    float* xn  = h   + (size_t)M_ * D_;             // M*D
    float* og  = xn  + (size_t)M_ * D_;             // M*D (also reused as h+concept)
    float* dcy = og  + (size_t)M_ * D_;             // M*F
    float* sp  = dcy + (size_t)M_ * F_;             // M*2F
    float* par = sp  + (size_t)M_ * 2 * F_;         // M*2*LAT
    float* acc = par + (size_t)M_ * 2 * LAT_;       // 1 (+pad)

    u16* bfb    = (u16*)(acc + 4);                  // bf16 region
    u16* xn_bf  = bfb;                              // M*D
    u16* u_bf   = xn_bf  + (size_t)M_ * D_;         // M*D
    u16* yc_bf  = u_bf   + (size_t)M_ * D_;         // M*2F
    u16* f1_bf  = yc_bf  + (size_t)M_ * 2 * F_;     // M*4D
    u16* par_bf = f1_bf  + (size_t)M_ * 4 * D_;     // M*2*LAT
    // bf16 weights (converted once per launch)
    u16* emb_bf = par_bf + (size_t)M_ * 2 * LAT_;   // V*D
    u16* gW_bf  = emb_bf + (size_t)V_ * D_;         // NL*D*D
    u16* oW_bf  = gW_bf  + (size_t)NL_ * D_ * D_;   // NL*D*D
    u16* dW_bf  = oW_bf  + (size_t)NL_ * D_ * D_;   // NL*F*D
    u16* siW_bf = dW_bf  + (size_t)NL_ * F_ * D_;   // NL*2F*D
    u16* soW_bf = siW_bf + (size_t)NL_ * 2 * F_ * D_; // NL*D*2F
    u16* w1_bf  = soW_bf + (size_t)NL_ * D_ * 2 * F_; // NL*4D*D
    u16* w2_bf  = w1_bf  + (size_t)NL_ * 4 * D_ * D_; // NL*D*4D
    u16* eW_bf  = w2_bf  + (size_t)NL_ * D_ * 4 * D_; // 2LAT*D
    u16* dcW_bf = eW_bf  + (size_t)2 * LAT_ * D_;     // D*LAT

    // -------- weight conversion (once per launch)
    launch_cvt(emb,        emb_bf, (size_t)V_ * D_,             stream);
    launch_cvt(gate_W,     gW_bf,  (size_t)NL_ * D_ * D_,       stream);
    launch_cvt(ogate_W,    oW_bf,  (size_t)NL_ * D_ * D_,       stream);
    launch_cvt(decay_W,    dW_bf,  (size_t)NL_ * F_ * D_,       stream);
    launch_cvt(spec_in_W,  siW_bf, (size_t)NL_ * 2 * F_ * D_,   stream);
    launch_cvt(spec_out_W, soW_bf, (size_t)NL_ * D_ * 2 * F_,   stream);
    launch_cvt(ffn_W1,     w1_bf,  (size_t)NL_ * 4 * D_ * D_,   stream);
    launch_cvt(ffn_W2,     w2_bf,  (size_t)NL_ * D_ * 4 * D_,   stream);
    launch_cvt(enc_W,      eW_bf,  (size_t)2 * LAT_ * D_,       stream);
    launch_cvt(dec_W,      dcW_bf, (size_t)D_ * LAT_,           stream);

    // h = emb[x]
    embed_k<<<M_, 256, 0, stream>>>(x, emb, h);

    for (int i = 0; i < NL_; ++i) {
        const u16* gW  = gW_bf  + (size_t)i * D_ * D_;
        const u16* oW  = oW_bf  + (size_t)i * D_ * D_;
        const u16* dW  = dW_bf  + (size_t)i * F_ * D_;
        const u16* siW = siW_bf + (size_t)i * 2 * F_ * D_;
        const u16* soW = soW_bf + (size_t)i * D_ * 2 * F_;
        const u16* w1  = w1_bf  + (size_t)i * 4 * D_ * D_;
        const u16* w2  = w2_bf  + (size_t)i * D_ * 4 * D_;

        // xn = LN(h)  (f32 + bf16)
        layernorm_k<<<M_, 256, 0, stream>>>(h, norm1_g + (size_t)i * D_,
                                            norm1_b + (size_t)i * D_, xn, xn_bf);
        // u = xn * sigmoid(xn @ gate_W^T + gb)   (bf16 only)
        launch_gemm(xn_bf, D_, gW, gate_b + (size_t)i * D_, xn, nullptr,
                    nullptr, u_bf, D_, D_, 1, 0, stream);
        // decay = sigmoid(u @ decay_W^T + db)
        launch_gemm(u_bf, D_, dW, decay_b + (size_t)i * F_, nullptr, nullptr,
                    dcy, nullptr, F_, D_, 1, 0, stream);
        // sp = u @ spec_in_W^T
        launch_gemm(u_bf, D_, siW, nullptr, nullptr, nullptr,
                    sp, nullptr, 2 * F_, D_, 0, 0, stream);
        // complex inclusive scan along T -> yc (bf16)
        scan_k<<<(B_ * F_ + 255) / 256, 256, 0, stream>>>(dcy, sp,
                                                          freq + (size_t)i * F_, yc_bf);
        // og = sigmoid(xn @ ogate_W^T + ob)
        launch_gemm(xn_bf, D_, oW, ogate_b + (size_t)i * D_, nullptr, nullptr,
                    og, nullptr, D_, D_, 1, 0, stream);
        // h = h + (yc @ spec_out_W^T) * og
        launch_gemm(yc_bf, 2 * F_, soW, nullptr, og, h,
                    h, nullptr, D_, 2 * F_, 0, 0, stream);
        // xn = LN(h) (ffn norm)
        layernorm_k<<<M_, 256, 0, stream>>>(h, ffn_g + (size_t)i * D_,
                                            ffn_b + (size_t)i * D_, nullptr, xn_bf);
        // f1 = gelu(xn @ W1^T + b1)   (bf16 only)
        launch_gemm(xn_bf, D_, w1, ffn_b1 + (size_t)i * 4 * D_, nullptr, nullptr,
                    nullptr, f1_bf, 4 * D_, D_, 2, 0, stream);
        // h = h + f1 @ W2^T + b2
        launch_gemm(f1_bf, 4 * D_, w2, ffn_b2 + (size_t)i * D_, nullptr, h,
                    h, nullptr, D_, 4 * D_, 0, 0, stream);
    }

    // VIB head
    layernorm_k<<<M_, 256, 0, stream>>>(h, vib_g, vib_b, nullptr, xn_bf);
    // params = xn @ enc_W^T + enc_b   (f32 for KL, bf16 for dec GEMM)
    launch_gemm(xn_bf, D_, eW_bf, enc_b, nullptr, nullptr,
                par, par_bf, 2 * LAT_, D_, 0, 0, stream);
    // og = h + gelu(mu @ dec_W^T + dec_b)    (mu = par[:, :LAT], lda = 2*LAT)
    launch_gemm(par_bf, 2 * LAT_, dcW_bf, dec_b, nullptr, h,
                og, nullptr, D_, LAT_, 2, 0, stream);
    // xn = LN(og)
    layernorm_k<<<M_, 256, 0, stream>>>(og, normf_g, normf_b, nullptr, xn_bf);
    // logits = xn @ emb^T   (streaming 524MB output: NT stores keep emb_bf in L2)
    launch_gemm(xn_bf, D_, emb_bf, nullptr, nullptr, nullptr,
                logits, nullptr, V_, D_, 0, 1, stream);

    // reg_loss
    init_acc_k<<<1, 1, 0, stream>>>(acc);
    kl_reduce_k<<<256, 256, 0, stream>>>(par, acc);
    kl_final_k<<<1, 1, 0, stream>>>(acc, reg);
}
